// SwinTransformer_40475771797561
// MI455X (gfx1250) — matrix operand order repaired and checked by reading, not compiled, not run on hardware
//
#include <hip/hip_runtime.h>
#include <math.h>

// ---------------- Types / bf16 helpers ----------------
typedef __bf16 bf16_t;
typedef __attribute__((ext_vector_type(16))) __bf16 v16bf;
typedef __attribute__((ext_vector_type(8)))  float  v8f;

__device__ __forceinline__ bf16_t f2bf(float f) {
  unsigned u = __builtin_bit_cast(unsigned, f);
  u += 0x7fffu + ((u >> 16) & 1u);               // round-to-nearest-even
  unsigned short h = (unsigned short)(u >> 16);
  return __builtin_bit_cast(bf16_t, h);
}
__device__ __forceinline__ float bf2f(bf16_t b) {
  unsigned short h = __builtin_bit_cast(unsigned short, b);
  unsigned u = ((unsigned)h) << 16;
  return __builtin_bit_cast(float, u);
}

__device__ __forceinline__ v8f wmma_bf16(v16bf a, v16bf b, v8f c) {
  // (neg_a, A, neg_b, B, c_mod, C, reuse_a, reuse_b)
  return __builtin_amdgcn_wmma_f32_16x16x32_bf16(false, a, false, b, (short)0, c,
                                                 false, false);
}

// ---------------- Problem constants ----------------
#define BB 16
#define HH 56
#define WW_ 56
#define CC 128
#define HEADS 4
#define HD 32
#define WS 7
#define NTOK 49
#define NW 64
#define TOK 50176          // BB*HH*WW_
#define PMROWS 12544       // BB*28*28

// ---------------- bf16 WMMA GEMM -----------------------------------------------
// Block = 256 threads = 8 waves; block owns a 128x16 output strip (8 M-tiles
// sharing one N-tile). Per 32-deep k-step the block stages the 32x16 f32 weight
// tile once (coalesced), converts to bf16, and stores it to LDS pre-swizzled in
// B-fragment order so each lane reads its fragment as contiguous 32B (ds b128).
// Double-buffered -> one barrier per k-step. Requires M % 128 == 0, N % 16 == 0,
// K % 32 == 0.
template <typename Epi>
__global__ __launch_bounds__(256) void gemm_bf16_kernel(
    const bf16_t* __restrict__ A, const float* __restrict__ W,
    int M, int N, int K, Epi epi) {
  __shared__ bf16_t wf[2][512];               // 2 x (32x16) fragment-ordered
  const int tid  = threadIdx.x;
  const int lane = tid & 31;
  const int wv   = tid >> 5;                  // wave in block: 0..7
  const int ntiles = N >> 4;
  const int strip  = blockIdx.x / ntiles;
  const int nt     = blockIdx.x - strip * ntiles;
  const int mt     = strip * 8 + wv;
  const int col = lane & 15;
  const int hi  = lane >> 4;

  v8f acc = {0.f, 0.f, 0.f, 0.f, 0.f, 0.f, 0.f, 0.f};
  const bf16_t* arow  = A + (size_t)(mt * 16 + col) * K;
  const float*  wbase = W + nt * 16;
  int buf = 0;
  for (int k0 = 0; k0 < K; k0 += 32) {
    // ---- stage 32x16 weight tile: 512 values, 2 per thread, coalesced ----
    {
      int i0 = tid, i1 = tid + 256;
      int ka = i0 >> 4, na = i0 & 15;
      int kb = i1 >> 4, nb = i1 & 15;
      float wa = wbase[(size_t)(k0 + ka) * N + na];
      float wbv = wbase[(size_t)(k0 + kb) * N + nb];
      // B-fragment order: lane = (k&16) + n, elem = k&15
      wf[buf][(((ka & 16)) + na) * 16 + (ka & 15)] = f2bf(wa);
      wf[buf][(((kb & 16)) + nb) * 16 + (kb & 15)] = f2bf(wbv);
    }
    __syncthreads();
    v16bf a, b;
    const bf16_t* pa = arow + k0 + hi * 16;
    if (k0 + 32 < K) __builtin_prefetch(arow + k0 + 32, 0, 1);
#pragma unroll
    for (int i = 0; i < 8; ++i) { a[i] = pa[i]; a[8 + i] = pa[8 + i]; }
    const bf16_t* pb = &wf[buf][lane * 16];   // contiguous 32B per lane
#pragma unroll
    for (int i = 0; i < 16; ++i) b[i] = pb[i];
    acc = wmma_bf16(a, b, acc);
    buf ^= 1;
  }
  const int n = nt * 16 + col;
  const int mbase = mt * 16 + hi * 8;
#pragma unroll
  for (int r = 0; r < 8; ++r) epi(mbase + r, n, acc[r]);
}

// ---------------- Epilogues ----------------
struct EpiQKV {                 // + bias, store bf16 (token,384)
  const float* bias; bf16_t* out;
  __device__ void operator()(int m, int n, float v) const {
    out[(size_t)m * 384 + n] = f2bf(v + bias[n]);
  }
};
struct EpiProj {                // + bias, window-reverse + unshift, residual x
  const float* bias; const float* x; float* x1;
  __device__ void operator()(int m, int n, float v) const {
    int win = m / NTOK, i = m - win * NTOK;
    int b = win >> 6, wi = win & 63, wh = wi >> 3, ww = wi & 7;
    int hs = wh * WS + i / WS + 3; if (hs >= HH) hs -= HH;
    int ws = ww * WS + i % WS + 3; if (ws >= WW_) ws -= WW_;
    size_t idx = (((size_t)b * HH + hs) * WW_ + ws) * CC + n;
    x1[idx] = x[idx] + v + bias[n];
  }
};
struct EpiGelu {                // + bias, exact GELU, store bf16 (token,512)
  const float* bias; bf16_t* out;
  __device__ void operator()(int m, int n, float v) const {
    float t = v + bias[n];
    float g = 0.5f * t * (1.0f + erff(t * 0.70710678118654752f));
    out[(size_t)m * 512 + n] = f2bf(g);
  }
};
struct EpiRes {                 // + bias, residual in-place: x2 = x1 + h
  const float* bias; float* x12;
  __device__ void operator()(int m, int n, float v) const {
    size_t idx = (size_t)m * CC + n;
    x12[idx] = x12[idx] + v + bias[n];
  }
};
struct EpiPlain {               // plain f32 store (patch-merge output)
  float* out; int ldo;
  __device__ void operator()(int m, int n, float v) const {
    out[(size_t)m * ldo + n] = v;
  }
};

// ---------------- LN + shift + window-partition -> bf16 A matrix ----------------
__global__ __launch_bounds__(256) void ln_shift_win_kernel(
    const float* __restrict__ x, const float* __restrict__ g,
    const float* __restrict__ bb, bf16_t* __restrict__ out) {
  int wave = blockIdx.x * (blockDim.x >> 5) + (threadIdx.x >> 5);
  int lane = threadIdx.x & 31;
  if (wave >= TOK) return;
  int win = wave / NTOK, i = wave - win * NTOK;
  int b = win >> 6, wi = win & 63, wh = wi >> 3, ww = wi & 7;
  int hs = wh * WS + i / WS + 3; if (hs >= HH) hs -= HH;
  int ws = ww * WS + i % WS + 3; if (ws >= WW_) ws -= WW_;
  const float* px = x + (((size_t)b * HH + hs) * WW_ + ws) * CC;
  float v[4], s = 0.f, s2 = 0.f;
#pragma unroll
  for (int j = 0; j < 4; ++j) { v[j] = px[lane * 4 + j]; s += v[j]; s2 += v[j] * v[j]; }
#pragma unroll
  for (int d = 1; d < 32; d <<= 1) { s += __shfl_xor(s, d, 32); s2 += __shfl_xor(s2, d, 32); }
  float mean = s * (1.0f / CC);
  float var  = s2 * (1.0f / CC) - mean * mean;
  float inv  = rsqrtf(var + 1.001e-05f);
  bf16_t* po = out + (size_t)wave * CC;
#pragma unroll
  for (int j = 0; j < 4; ++j) {
    int c = lane * 4 + j;
    po[c] = f2bf((v[j] - mean) * inv * g[c] + bb[c]);
  }
}

// ---------------- LN2 (natural token order) -> bf16 ----------------
__global__ __launch_bounds__(256) void ln2_kernel(
    const float* __restrict__ x1, const float* __restrict__ g,
    const float* __restrict__ bb, bf16_t* __restrict__ out) {
  int wave = blockIdx.x * (blockDim.x >> 5) + (threadIdx.x >> 5);
  int lane = threadIdx.x & 31;
  if (wave >= TOK) return;
  const float* px = x1 + (size_t)wave * CC;
  float v[4], s = 0.f, s2 = 0.f;
#pragma unroll
  for (int j = 0; j < 4; ++j) { v[j] = px[lane * 4 + j]; s += v[j]; s2 += v[j] * v[j]; }
#pragma unroll
  for (int d = 1; d < 32; d <<= 1) { s += __shfl_xor(s, d, 32); s2 += __shfl_xor(s2, d, 32); }
  float mean = s * (1.0f / CC);
  float var  = s2 * (1.0f / CC) - mean * mean;
  float inv  = rsqrtf(var + 1.001e-05f);
  bf16_t* po = out + (size_t)wave * CC;
#pragma unroll
  for (int j = 0; j < 4; ++j) {
    int c = lane * 4 + j;
    po[c] = f2bf((v[j] - mean) * inv * g[c] + bb[c]);
  }
}

// ---------------- Fused windowed attention (one window-head per wave32) --------
__device__ __forceinline__ int regionId(int r, int c) {
  int gr = (r < 49) ? 0 : ((r < 53) ? 1 : 2);
  int gc = (c < 49) ? 0 : ((c < 53) ? 1 : 2);
  return gr * 3 + gc;
}

__global__ __launch_bounds__(32) void attn_kernel(
    const bf16_t* __restrict__ qkv, const float* __restrict__ bias_table,
    bf16_t* __restrict__ attnout) {
  __shared__ bf16_t probs[64 * 64];
  const int lane = threadIdx.x;
  const int head = blockIdx.x & 3;
  const int win  = blockIdx.x >> 2;
  const int wi = win & 63, wh = wi >> 3, ww = wi & 7;
  const size_t tok0 = (size_t)win * NTOK;
  const float scale = 0.17677669529663687f;   // 1/sqrt(32)
  const int col = lane & 15;
  const int hi  = lane >> 4;

  // ---- Phase 1: scores (Q@K^T) + bias + mask + softmax -> LDS (bf16, 64x64) ----
  for (int mt = 0; mt < 4; ++mt) {
    v8f acc[4];
#pragma unroll
    for (int nt = 0; nt < 4; ++nt) acc[nt] = (v8f){0.f,0.f,0.f,0.f,0.f,0.f,0.f,0.f};
    v16bf a;
    {
      const bf16_t* qp = qkv + (tok0 + mt * 16 + col) * 384 + head * HD + hi * 16;
#pragma unroll
      for (int i = 0; i < 8; ++i) { a[i] = qp[i]; a[8 + i] = qp[8 + i]; }
    }
#pragma unroll
    for (int nt = 0; nt < 4; ++nt) {
      v16bf bfrag;
      const bf16_t* kp = qkv + (tok0 + nt * 16 + col) * 384 + 128 + head * HD + hi * 16;
#pragma unroll
      for (int i = 0; i < 16; ++i) bfrag[i] = kp[i];
      acc[nt] = wmma_bf16(a, bfrag, acc[nt]);
    }
    float sc[4][8];
#pragma unroll
    for (int nt = 0; nt < 4; ++nt) {
      int kcol = nt * 16 + col;
#pragma unroll
      for (int r = 0; r < 8; ++r) {
        int qrow = mt * 16 + hi * 8 + r;
        float v = acc[nt][r] * scale;
        if (kcol < NTOK && qrow < NTOK) {
          int fxq = qrow / WS, fyq = qrow % WS, fxk = kcol / WS, fyk = kcol % WS;
          int rel = (fxq - fxk + 6) * 13 + (fyq - fyk + 6);
          v += bias_table[rel * HEADS + head];
          int gq = regionId(wh * WS + fxq, ww * WS + fyq);
          int gk = regionId(wh * WS + fxk, ww * WS + fyk);
          if (gq != gk) v += -100.0f;
        } else {
          v = -1e30f;
        }
        sc[nt][r] = v;
      }
    }
    // row-wise softmax: each row lives on 16 lanes (fixed lane-half), xor-reduce
#pragma unroll
    for (int r = 0; r < 8; ++r) {
      float mx = fmaxf(fmaxf(sc[0][r], sc[1][r]), fmaxf(sc[2][r], sc[3][r]));
#pragma unroll
      for (int d = 1; d < 16; d <<= 1) mx = fmaxf(mx, __shfl_xor(mx, d, 32));
      float sm = 0.f;
#pragma unroll
      for (int nt = 0; nt < 4; ++nt) { sc[nt][r] = __expf(sc[nt][r] - mx); sm += sc[nt][r]; }
#pragma unroll
      for (int d = 1; d < 16; d <<= 1) sm += __shfl_xor(sm, d, 32);
      float invs = 1.0f / sm;
      int row = mt * 16 + hi * 8 + r;
#pragma unroll
      for (int nt = 0; nt < 4; ++nt)
        probs[row * 64 + nt * 16 + col] = f2bf(sc[nt][r] * invs);
    }
  }
  __syncthreads();

  // ---- Phase 2: out = probs @ V  (64x64 @ 64x32) ----
  for (int mt = 0; mt < 4; ++mt) {
    v8f acc2[2];
    acc2[0] = (v8f){0.f,0.f,0.f,0.f,0.f,0.f,0.f,0.f};
    acc2[1] = (v8f){0.f,0.f,0.f,0.f,0.f,0.f,0.f,0.f};
#pragma unroll
    for (int kk = 0; kk < 64; kk += 32) {
      v16bf a;
      const bf16_t* pp = &probs[(mt * 16 + col) * 64 + kk + hi * 16];
#pragma unroll
      for (int i = 0; i < 8; ++i) { a[i] = pp[i]; a[8 + i] = pp[8 + i]; }
#pragma unroll
      for (int nt = 0; nt < 2; ++nt) {
        v16bf bfrag;
        int ch = nt * 16 + col;
        const bf16_t* vp = qkv + (tok0 + kk + hi * 16) * 384 + 256 + head * HD + ch;
#pragma unroll
        for (int i = 0; i < 16; ++i) bfrag[i] = vp[(size_t)i * 384];
        acc2[nt] = wmma_bf16(a, bfrag, acc2[nt]);
      }
    }
#pragma unroll
    for (int nt = 0; nt < 2; ++nt) {
#pragma unroll
      for (int r = 0; r < 8; ++r) {
        int qrow = mt * 16 + hi * 8 + r;
        if (qrow < NTOK)
          attnout[(tok0 + qrow) * CC + head * HD + nt * 16 + col] = f2bf(acc2[nt][r]);
      }
    }
  }
}

// ---------------- Patch-merge gather + channel permute + LN -> bf16 ------------
__global__ __launch_bounds__(256) void pm_gather_ln_kernel(
    const float* __restrict__ x2, const float* __restrict__ g,
    const float* __restrict__ bb, bf16_t* __restrict__ out) {
  int wave = blockIdx.x * (blockDim.x >> 5) + (threadIdx.x >> 5);
  int lane = threadIdx.x & 31;
  if (wave >= PMROWS) return;
  int b = wave / 784, rem = wave - b * 784;
  int i2 = rem / 28, j2 = rem - i2 * 28;
  float vals[16], s = 0.f, s2 = 0.f;
#pragma unroll
  for (int j = 0; j < 16; ++j) {
    int c = lane * 16 + j;
    int blk = c >> 7, cc = c & 127;
    int dh = blk & 1, dw = (blk >> 1) & 1;       // order: x00, x10, x01, x11
    size_t idx = (((size_t)b * HH + 2 * i2 + dh) * WW_ + 2 * j2 + dw) * CC + cc;
    float v = x2[idx];
    vals[j] = v; s += v; s2 += v * v;
  }
#pragma unroll
  for (int d = 1; d < 32; d <<= 1) { s += __shfl_xor(s, d, 32); s2 += __shfl_xor(s2, d, 32); }
  float mean = s * (1.0f / 512.0f);
  float var  = s2 * (1.0f / 512.0f) - mean * mean;
  float inv  = rsqrtf(var + 1.001e-05f);
  bf16_t* po = out + (size_t)wave * 512;
#pragma unroll
  for (int j = 0; j < 16; ++j) {
    int c = lane * 16 + j;
    po[c] = f2bf((vals[j] - mean) * inv * g[c] + bb[c]);
  }
}

// ---------------- Launch ----------------
extern "C" void kernel_launch(void* const* d_in, const int* in_sizes, int n_in,
                              void* d_out, int out_size, void* d_ws, size_t ws_size,
                              hipStream_t stream) {
  const float* x          = (const float*)d_in[0];
  const float* ln1_g      = (const float*)d_in[1];
  const float* ln1_b      = (const float*)d_in[2];
  const float* qkv_w      = (const float*)d_in[3];
  const float* qkv_b      = (const float*)d_in[4];
  const float* proj_w     = (const float*)d_in[5];
  const float* proj_b     = (const float*)d_in[6];
  const float* bias_table = (const float*)d_in[7];
  const float* ln2_g      = (const float*)d_in[8];
  const float* ln2_b      = (const float*)d_in[9];
  const float* mlp_w1     = (const float*)d_in[10];
  const float* mlp_b1     = (const float*)d_in[11];
  const float* mlp_w2     = (const float*)d_in[12];
  const float* mlp_b2     = (const float*)d_in[13];
  const float* pm_ln_g    = (const float*)d_in[14];
  const float* pm_ln_b    = (const float*)d_in[15];
  const float* pm_w       = (const float*)d_in[16];
  float* out = (float*)d_out;

  // Workspace layout (bytes), with deliberate reuse:
  //  [0, 12.8M)     winA -> later h_bf -> later pm_in     (each exactly 12,845,056)
  //  [12.8M, 64.2M) qkv(38,535,168)+attnout(12,845,056) -> later h1 (51,380,224)
  //  [64.2M, 89.9M) x1 f32 (in-place becomes x2)
  char* ws = (char*)d_ws;
  const size_t SZ_WINA = (size_t)TOK * CC * 2;
  const size_t SZ_QKV  = (size_t)TOK * 384 * 2;
  bf16_t* winA    = (bf16_t*)(ws);
  bf16_t* qkvbuf  = (bf16_t*)(ws + SZ_WINA);
  bf16_t* attnout = (bf16_t*)(ws + SZ_WINA + SZ_QKV);
  bf16_t* h_bf    = (bf16_t*)(ws);                 // reuse winA
  bf16_t* h1      = (bf16_t*)(ws + SZ_WINA);       // reuse qkv+attnout
  bf16_t* pm_in   = (bf16_t*)(ws);                 // reuse h_bf
  float*  x1      = (float*)(ws + SZ_WINA + SZ_QKV + (size_t)TOK * CC * 2);

  const int WPB = 8;  // waves per 256-thread block
  // 1) LN1 + shift + window partition
  ln_shift_win_kernel<<<TOK / WPB, 256, 0, stream>>>(x, ln1_g, ln1_b, winA);
  // 2) QKV GEMM: (50176x128)@(128x384); grid = (M/128)*(N/16) blocks
  gemm_bf16_kernel<<<(TOK / 128) * (384 / 16), 256, 0, stream>>>(
      winA, qkv_w, TOK, 384, CC, EpiQKV{qkv_b, qkvbuf});
  // 3) fused windowed attention (scores+bias+mask+softmax+AV), 1 wave / window-head
  attn_kernel<<<BB * NW * HEADS, 32, 0, stream>>>(qkvbuf, bias_table, attnout);
  // 4) proj GEMM + window-reverse + unshift + residual1 -> x1
  gemm_bf16_kernel<<<(TOK / 128) * (CC / 16), 256, 0, stream>>>(
      attnout, proj_w, TOK, CC, CC, EpiProj{proj_b, x, x1});
  // 5) LN2
  ln2_kernel<<<TOK / WPB, 256, 0, stream>>>(x1, ln2_g, ln2_b, h_bf);
  // 6) MLP fc1 + GELU: (50176x128)@(128x512)
  gemm_bf16_kernel<<<(TOK / 128) * (512 / 16), 256, 0, stream>>>(
      h_bf, mlp_w1, TOK, 512, CC, EpiGelu{mlp_b1, h1});
  // 7) MLP fc2 + residual2 (in-place on x1 -> x2)
  gemm_bf16_kernel<<<(TOK / 128) * (CC / 16), 256, 0, stream>>>(
      h1, mlp_w2, TOK, CC, 512, EpiRes{mlp_b2, x1});
  // 8) patch-merge gather + permute + LN
  pm_gather_ln_kernel<<<PMROWS / WPB, 256, 0, stream>>>(x1, pm_ln_g, pm_ln_b, pm_in);
  // 9) final GEMM: (12544x512)@(512x256) -> d_out
  gemm_bf16_kernel<<<(PMROWS / 128) * (256 / 16), 256, 0, stream>>>(
      pm_in, pm_w, PMROWS, 256, 512, EpiPlain{out, 256});
}
